// CausalSelfAttention_75943611728281
// MI455X (gfx1250) — matrix operand order repaired and checked by reading, not compiled, not run on hardware
//
#include <hip/hip_runtime.h>
#include <hip/hip_bf16.h>

typedef __bf16 bf16_t;
typedef bf16_t v16bf __attribute__((ext_vector_type(16)));
typedef bf16_t v8bf  __attribute__((ext_vector_type(8)));
typedef float  v8f   __attribute__((ext_vector_type(8)));
typedef unsigned int u32;
typedef unsigned int u32x4 __attribute__((ext_vector_type(4)));
typedef int i32x4 __attribute__((ext_vector_type(4)));
typedef int i32x8 __attribute__((ext_vector_type(8)));

#define LDP 72  // padded LDS row stride in elements (144B): 32 dwords data + 4 dwords pad

// ---------------------------------------------------------------------------
// DPP16 butterfly reductions within each 16-lane row (no LDS, no waits).
// Patterns: xor1 = quad_perm(1,0,3,2)=0xB1, xor2 = quad_perm(2,3,0,1)=0x4E,
// then row_half_mirror (0x141) and row_mirror (0x140) complete the tree.
// ---------------------------------------------------------------------------
template <int CTRL>
__device__ __forceinline__ float dpp_movf(float v) {
  int r = __builtin_amdgcn_update_dpp(0, __builtin_bit_cast(int, v), CTRL, 0xf,
                                      0xf, true);
  return __builtin_bit_cast(float, r);
}
__device__ __forceinline__ float row16_max(float x) {
  x = fmaxf(x, dpp_movf<0xB1>(x));
  x = fmaxf(x, dpp_movf<0x4E>(x));
  x = fmaxf(x, dpp_movf<0x141>(x));
  x = fmaxf(x, dpp_movf<0x140>(x));
  return x;
}
__device__ __forceinline__ float row16_sum(float x) {
  x += dpp_movf<0xB1>(x);
  x += dpp_movf<0x4E>(x);
  x += dpp_movf<0x141>(x);
  x += dpp_movf<0x140>(x);
  return x;
}

// ---------------------------------------------------------------------------
// TDM: async DMA of a 2-D bf16 tile (tile_d0 x tile_d1, row stride stride0
// elements) from global memory into LDS at byte offset lds_off, with D# pad
// fields inserting 4 dwords of padding every 32 dwords (64 bf16 -> LDP=72).
// One instruction per wave; tracked by TENSORcnt. 6-arg toolchain builtin.
// ---------------------------------------------------------------------------
__device__ __forceinline__ void tdm_load_2d_bf16(
    unsigned lds_off, const bf16_t* gptr, int tile_d0, int tile_d1,
    long long stride0_elems, int tensor_d0, int tensor_d1) {
  unsigned long long ga = (unsigned long long)(uintptr_t)gptr;
  u32x4 g0;
  g0[0] = 1u;                                              // count=1, user mode
  g0[1] = lds_off;                                         // lds_addr
  g0[2] = (u32)(ga & 0xffffffffull);                       // global_addr[31:0]
  g0[3] = (u32)((ga >> 32) & 0x01ffffffull) | (2u << 30);  // addr[56:32] | type=2

  unsigned long long s0 = (unsigned long long)stride0_elems;  // 48-bit field
  i32x8 g1;
  // data_size=1 (2B) | pad_enable | pad_interval=4 (32 dwords) | pad_amount=3 (4 dwords)
  g1[0] = (int)((1u << 16) | (1u << 20) | (4u << 22) | (3u << 25));
  g1[1] = (int)(((u32)tensor_d0 & 0xffffu) << 16);                       // dim0[15:0]
  g1[2] = (int)(((u32)tensor_d0 >> 16) | (((u32)tensor_d1 & 0xffffu) << 16));
  g1[3] = (int)(((u32)tensor_d1 >> 16) | (((u32)tile_d0 & 0xffffu) << 16));
  g1[4] = (int)((u32)tile_d1 & 0xffffu);                                 // tile_dim2=0
  g1[5] = (int)(u32)(s0 & 0xffffffffull);                                // stride0 lo
  g1[6] = (int)(u32)((s0 >> 32) & 0xffffull);                            // stride0 hi | stride1=0
  g1[7] = 0;
  i32x4 zz4 = {0, 0, 0, 0};
  i32x8 zz8 = {0, 0, 0, 0, 0, 0, 0, 0};
  __builtin_amdgcn_tensor_load_to_lds(g0, g1, zz4, zz4, zz8, 0);
}

// ---------------------------------------------------------------------------
// f32 -> bf16 conversion (grid-stride)
// ---------------------------------------------------------------------------
__global__ void convert_f32_to_bf16(const float* __restrict__ in,
                                    bf16_t* __restrict__ out, size_t n) {
  size_t i = (size_t)blockIdx.x * blockDim.x + threadIdx.x;
  size_t stride = (size_t)gridDim.x * blockDim.x;
  for (; i < n; i += stride) out[i] = (bf16_t)in[i];
}

// Transpose+convert: in f32 [K][N] -> out bf16 [N][K] (coalesced writes)
__global__ void transpose_convert_f32_bf16(const float* __restrict__ in,
                                           bf16_t* __restrict__ out, int K,
                                           int N) {
  size_t total = (size_t)K * N;
  size_t i = (size_t)blockIdx.x * blockDim.x + threadIdx.x;
  size_t stride = (size_t)gridDim.x * blockDim.x;
  for (; i < total; i += stride) {
    size_t n = i / K, k = i % K;
    out[i] = (bf16_t)in[k * (size_t)N + n];
  }
}

// ---------------------------------------------------------------------------
// bf16 GEMM via WMMA, TDM-staged + double-buffered LDS panels.
//   A [M,K] row-major bf16, Bt [N,K] row-major bf16 (pre-transposed weights).
// Macro-tile 128x128, K-step 64. 8 waves, each 64x32 (4x2 WMMA tiles).
// MODE 0: Cout = f32 [M,N] (+bias).   MODE 1: QKV split -> qk bf16 [M,2048]
// for cols<2048; cols>=2048 go transposed to vt bf16 [B*H, 64, 2048] (+bias).
// ---------------------------------------------------------------------------
#define PANEL_B (128 * LDP * 2)  // 18432 bytes per panel

template <int MODE>
__global__ __launch_bounds__(256) void gemm_bf16(
    const bf16_t* __restrict__ A, const bf16_t* __restrict__ Bt,
    const float* __restrict__ bias, void* __restrict__ Cout,
    bf16_t* __restrict__ vt, int M, int N, int K) {
  extern __shared__ char smem[];  // A0 A1 B0 B1 panels
  const unsigned ldsbase = __builtin_amdgcn_groupstaticsize();

  const int tid  = threadIdx.x;
  const int lane = tid & 31;
  const int wave = tid >> 5;
  const int hi   = (lane >> 4) & 1;
  const int ln   = lane & 15;

  const int m0 = blockIdx.y * 128;
  const int n0 = blockIdx.x * 128;
  const int wm = (wave & 1) * 64;
  const int wn = (wave >> 1) * 32;

  v8f zero = {0.f, 0.f, 0.f, 0.f, 0.f, 0.f, 0.f, 0.f};
  v8f acc[4][2];
#pragma unroll
  for (int rt = 0; rt < 4; ++rt)
#pragma unroll
    for (int ct = 0; ct < 2; ++ct) acc[rt][ct] = zero;

  // Prologue: wave 0 issues TDM for first K panel pair into buffer 0.
  if (wave == 0) {
    tdm_load_2d_bf16(ldsbase + 0 * PANEL_B, A + (size_t)m0 * K, 64, 128, K, 64, 128);
    tdm_load_2d_bf16(ldsbase + 2 * PANEL_B, Bt + (size_t)n0 * K, 64, 128, K, 64, 128);
  }

  int cur = 0;
  for (int k0 = 0; k0 < K; k0 += 64) {
    __syncthreads();  // alt buffer free (prev iteration's reads done)
    if (wave == 0) {
      if (k0 + 64 < K) {
        int nxt = cur ^ 1;
        tdm_load_2d_bf16(ldsbase + nxt * PANEL_B, A + (size_t)m0 * K + k0 + 64,
                         64, 128, K, 64, 128);
        tdm_load_2d_bf16(ldsbase + (2 + nxt) * PANEL_B,
                         Bt + (size_t)n0 * K + k0 + 64, 64, 128, K, 64, 128);
        __builtin_amdgcn_s_wait_tensorcnt(2);  // current pair landed (in-order)
      } else {
        __builtin_amdgcn_s_wait_tensorcnt(0);
      }
    }
    __syncthreads();  // current buffer visible to all waves

    const bf16_t* Alds = (const bf16_t*)(smem + cur * PANEL_B);
    const bf16_t* Blds = (const bf16_t*)(smem + (2 + cur) * PANEL_B);

#pragma unroll
    for (int ks = 0; ks < 2; ++ks) {
      v16bf a[4], b[2];
#pragma unroll
      for (int rt = 0; rt < 4; ++rt) {
        // A-frag: lane<16 -> row base+ln, K = ks*32 + {0..7,16..23}; hi lanes +8
        const bf16_t* p = Alds + (wm + rt * 16 + ln) * LDP + ks * 32 + hi * 8;
        v8bf lo = *(const v8bf*)(p);
        v8bf hh = *(const v8bf*)(p + 16);
        a[rt] = __builtin_shufflevector(lo, hh, 0, 1, 2, 3, 4, 5, 6, 7, 8, 9,
                                        10, 11, 12, 13, 14, 15);
      }
#pragma unroll
      for (int ct = 0; ct < 2; ++ct) {
        // B-frag: lane n holds column n, K = ks*32 + {0..7,16..23}; hi lanes +8
        const bf16_t* p = Blds + (wn + ct * 16 + ln) * LDP + ks * 32 + hi * 8;
        v8bf lo = *(const v8bf*)(p);
        v8bf hh = *(const v8bf*)(p + 16);
        b[ct] = __builtin_shufflevector(lo, hh, 0, 1, 2, 3, 4, 5, 6, 7, 8, 9,
                                        10, 11, 12, 13, 14, 15);
      }
#pragma unroll
      for (int rt = 0; rt < 4; ++rt)
#pragma unroll
        for (int ct = 0; ct < 2; ++ct)
          acc[rt][ct] = __builtin_amdgcn_wmma_f32_16x16x32_bf16(
              false, a[rt], false, b[ct], (short)0, acc[rt][ct], false, false);
    }
    cur ^= 1;
  }

  // Epilogue: C layout -> lane holds col = ln, rows r + 8*hi
#pragma unroll
  for (int ct = 0; ct < 2; ++ct) {
    int col = n0 + wn + ct * 16 + ln;
    float bb = bias ? bias[col] : 0.f;
#pragma unroll
    for (int rt = 0; rt < 4; ++rt) {
#pragma unroll
      for (int r = 0; r < 8; ++r) {
        int row = m0 + wm + rt * 16 + r + hi * 8;
        float vout = acc[rt][ct][r] + bb;
        if constexpr (MODE == 0) {
          ((float*)Cout)[(size_t)row * N + col] = vout;
        } else {
          if (col < 2048) {
            ((bf16_t*)Cout)[(size_t)row * 2048 + col] = (bf16_t)vout;
          } else {
            int hh2 = (col - 2048) >> 6, d = (col - 2048) & 63;
            int bb2 = row >> 11, s = row & 2047;  // S = 2048
            vt[((((size_t)bb2 * 16 + hh2) * 64) + d) * 2048 + s] = (bf16_t)vout;
          }
        }
      }
    }
  }
}

// ---------------------------------------------------------------------------
// Fused causal flash attention with TDM-staged, double-buffered K / Vt tiles.
// qk: [B*S, 2048] bf16 (q | k per head at col h*64 / 1024+h*64)
// vt: [B*H, 64, S] bf16 (V transposed: row d, contiguous keys)
// Y : [B*S, 1024] bf16
// Grid (S/128, B*H). 8 waves x 16 query rows.
// ---------------------------------------------------------------------------
#define KV_B (64 * LDP * 2)  // 9216 bytes per K/V tile

__global__ __launch_bounds__(256) void flash_attn(
    const bf16_t* __restrict__ qk, const bf16_t* __restrict__ vt,
    bf16_t* __restrict__ Y) {
  constexpr int S = 2048, DH = 64, C = 1024;
  constexpr float scale = 0.125f;  // 1/sqrt(64)
  extern __shared__ char smem[];   // K0 K1 Vt0 Vt1 | P (8 waves x 16 x LDP)
  const unsigned ldsbase = __builtin_amdgcn_groupstaticsize();

  const int tid  = threadIdx.x;
  const int lane = tid & 31;
  const int wave = tid >> 5;
  const int hi   = (lane >> 4) & 1;
  const int ln   = lane & 15;

  const int bh = blockIdx.y;
  const int b  = bh >> 4;
  const int h  = bh & 15;
  const int q0 = blockIdx.x * 128;
  const int qw = q0 + wave * 16;

  const bf16_t* qbase = qk + (size_t)b * S * 2048 + h * DH;
  const bf16_t* kbase = qbase + 1024;
  const bf16_t* vbase = vt + (size_t)bh * DH * S;  // [d][s]

  // Q fragments from global (A layout)
  v16bf qf[2];
#pragma unroll
  for (int ks = 0; ks < 2; ++ks) {
    const bf16_t* p = qbase + (size_t)(qw + ln) * 2048 + ks * 32 + hi * 8;
    v8bf lo = *(const v8bf*)(p);
    v8bf hh = *(const v8bf*)(p + 16);
    qf[ks] = __builtin_shufflevector(lo, hh, 0, 1, 2, 3, 4, 5, 6, 7, 8, 9, 10,
                                     11, 12, 13, 14, 15);
  }

  v8f zero = {0.f, 0.f, 0.f, 0.f, 0.f, 0.f, 0.f, 0.f};
  v8f o[4];
#pragma unroll
  for (int t = 0; t < 4; ++t) o[t] = zero;
  float m_i[8], l_i[8];
#pragma unroll
  for (int r = 0; r < 8; ++r) { m_i[r] = -1e30f; l_i[r] = 0.f; }

  const int ktiles = blockIdx.x * 2 + 2;  // keys up to q0+128

  // Prologue: wave 0 issues TDM for tile 0 (K row-major, Vt row-major)
  if (wave == 0) {
    tdm_load_2d_bf16(ldsbase + 0 * KV_B, kbase, 64, 64, 2048, 64, 64);
    tdm_load_2d_bf16(ldsbase + 2 * KV_B, vbase, 64, 64, S, 64, 64);
  }

  int cur = 0;
  for (int kt = 0; kt < ktiles; ++kt) {
    const int key0 = kt * 64;
    __syncthreads();
    if (wave == 0) {
      if (kt + 1 < ktiles) {
        int nxt = cur ^ 1;
        tdm_load_2d_bf16(ldsbase + nxt * KV_B,
                         kbase + (size_t)(key0 + 64) * 2048, 64, 64, 2048, 64, 64);
        tdm_load_2d_bf16(ldsbase + (2 + nxt) * KV_B, vbase + key0 + 64, 64, 64,
                         S, 64, 64);
        __builtin_amdgcn_s_wait_tensorcnt(2);
      } else {
        __builtin_amdgcn_s_wait_tensorcnt(0);
      }
    }
    __syncthreads();

    const bf16_t* Klds  = (const bf16_t*)(smem + cur * KV_B);
    const bf16_t* Vtlds = (const bf16_t*)(smem + (2 + cur) * KV_B);
    bf16_t* Plds = (bf16_t*)(smem + 4 * KV_B) + wave * 16 * LDP;

    // Scores: 16x64 = Q(16x64) @ K^T; B-frag col n = key, K-dim = head dim
    v8f sc[4];
#pragma unroll
    for (int ct = 0; ct < 4; ++ct) sc[ct] = zero;
#pragma unroll
    for (int ks = 0; ks < 2; ++ks) {
#pragma unroll
      for (int ct = 0; ct < 4; ++ct) {
        const bf16_t* p = Klds + (ct * 16 + ln) * LDP + ks * 32 + hi * 8;
        v8bf lo = *(const v8bf*)(p);
        v8bf hh = *(const v8bf*)(p + 16);
        v16bf bfr = __builtin_shufflevector(lo, hh, 0, 1, 2, 3, 4, 5, 6, 7, 8,
                                            9, 10, 11, 12, 13, 14, 15);
        sc[ct] = __builtin_amdgcn_wmma_f32_16x16x32_bf16(
            false, qf[ks], false, bfr, (short)0, sc[ct], false, false);
      }
    }

    // Online softmax per row (row = r + 8*hi, spread across 16 lanes).
    // Row reductions are DPP16 butterflies (pure VALU, no LDS round-trips).
    float alpha[8];
#pragma unroll
    for (int r = 0; r < 8; ++r) {
      const int qrow = qw + r + hi * 8;
      float mx = -1e30f;
#pragma unroll
      for (int ct = 0; ct < 4; ++ct) {
        float s = sc[ct][r] * scale;
        if (key0 + ct * 16 + ln > qrow) s = -1e30f;  // causal mask
        sc[ct][r] = s;
        mx = fmaxf(mx, s);
      }
      mx = row16_max(mx);
      const float mnew = fmaxf(m_i[r], mx);
      const float al = __expf(m_i[r] - mnew);
      float rowsum = 0.f;
#pragma unroll
      for (int ct = 0; ct < 4; ++ct) {
        float pv = __expf(sc[ct][r] - mnew);
        sc[ct][r] = pv;
        rowsum += pv;
      }
      rowsum = row16_sum(rowsum);
      l_i[r] = l_i[r] * al + rowsum;
      m_i[r] = mnew;
      alpha[r] = al;
    }
#pragma unroll
    for (int t = 0; t < 4; ++t)
#pragma unroll
      for (int r = 0; r < 8; ++r) o[t][r] *= alpha[r];

    // P (C-layout) -> per-wave LDS (same-wave DS ordering, no barrier needed)
#pragma unroll
    for (int ct = 0; ct < 4; ++ct)
#pragma unroll
      for (int r = 0; r < 8; ++r)
        Plds[(r + hi * 8) * LDP + ct * 16 + ln] = (bf16_t)sc[ct][r];

    // Reload P as A-fragments; O += P @ V (B-frag from Vt[d][key])
    v16bf pf[2];
#pragma unroll
    for (int ks = 0; ks < 2; ++ks) {
      const bf16_t* p = Plds + ln * LDP + ks * 32 + hi * 8;
      v8bf lo = *(const v8bf*)(p);
      v8bf hh = *(const v8bf*)(p + 16);
      pf[ks] = __builtin_shufflevector(lo, hh, 0, 1, 2, 3, 4, 5, 6, 7, 8, 9,
                                       10, 11, 12, 13, 14, 15);
    }
#pragma unroll
    for (int ks = 0; ks < 2; ++ks) {
#pragma unroll
      for (int dt = 0; dt < 4; ++dt) {
        const bf16_t* p = Vtlds + (dt * 16 + ln) * LDP + ks * 32 + hi * 8;
        v8bf lo = *(const v8bf*)(p);
        v8bf hh = *(const v8bf*)(p + 16);
        v16bf bfr = __builtin_shufflevector(lo, hh, 0, 1, 2, 3, 4, 5, 6, 7, 8,
                                            9, 10, 11, 12, 13, 14, 15);
        o[dt] = __builtin_amdgcn_wmma_f32_16x16x32_bf16(
            false, pf[ks], false, bfr, (short)0, o[dt], false, false);
      }
    }
    cur ^= 1;
  }

  // Finalize: y = O / l
#pragma unroll
  for (int dt = 0; dt < 4; ++dt) {
#pragma unroll
    for (int r = 0; r < 8; ++r) {
      const int row = qw + r + hi * 8;
      float val = o[dt][r] / l_i[r];
      Y[(size_t)(b * S + row) * C + h * DH + dt * 16 + ln] = (bf16_t)val;
    }
  }
}

// ---------------------------------------------------------------------------
static inline size_t align_up(size_t x) { return (x + 255) & ~(size_t)255; }

extern "C" void kernel_launch(void* const* d_in, const int* in_sizes, int n_in,
                              void* d_out, int out_size, void* d_ws,
                              size_t ws_size, hipStream_t stream) {
  const float* x      = (const float*)d_in[0];
  const float* w_qkv  = (const float*)d_in[1];
  const float* b_qkv  = (const float*)d_in[2];
  const float* w_proj = (const float*)d_in[3];
  const float* b_proj = (const float*)d_in[4];
  float* out = (float*)d_out;

  const int B = 4, S = 2048, C = 1024, C3 = 3072, H = 16;
  const size_t M = (size_t)B * S;  // 8192

  char* ws = (char*)d_ws;
  bf16_t* x_bf    = (bf16_t*)ws;  ws += align_up(M * C * sizeof(bf16_t));
  bf16_t* wqkvT   = (bf16_t*)ws;  ws += align_up((size_t)C3 * C * sizeof(bf16_t));
  bf16_t* wprojT  = (bf16_t*)ws;  ws += align_up((size_t)C * C * sizeof(bf16_t));
  bf16_t* qk_bf   = (bf16_t*)ws;  ws += align_up(M * 2048 * sizeof(bf16_t));
  bf16_t* vt_bf   = (bf16_t*)ws;  ws += align_up((size_t)B * H * 64 * S * sizeof(bf16_t));
  bf16_t* y_bf    = (bf16_t*)ws;  ws += align_up(M * C * sizeof(bf16_t));

  // 1) conversions: x straight; weights transposed to [N][K]
  convert_f32_to_bf16<<<dim3(2048), dim3(256), 0, stream>>>(x, x_bf, M * C);
  transpose_convert_f32_bf16<<<dim3(4096), dim3(256), 0, stream>>>(w_qkv, wqkvT, C, C3);
  transpose_convert_f32_bf16<<<dim3(2048), dim3(256), 0, stream>>>(w_proj, wprojT, C, C);

  // 2) QKV projection (q,k -> qk_bf; v -> vt_bf transposed)
  {
    dim3 grid(C3 / 128, (int)(M / 128));
    gemm_bf16<1><<<grid, dim3(256), 4 * PANEL_B, stream>>>(
        x_bf, wqkvT, b_qkv, (void*)qk_bf, vt_bf, (int)M, C3, C);
  }
  // 3) fused causal flash attention
  {
    dim3 grid(S / 128, B * H);
    size_t lds = 4 * KV_B + 8 * 16 * LDP * sizeof(bf16_t);
    flash_attn<<<grid, dim3(256), lds, stream>>>(qk_bf, vt_bf, y_bf);
  }
  // 4) output projection -> f32 out
  {
    dim3 grid(C / 128, (int)(M / 128));
    gemm_bf16<0><<<grid, dim3(256), 4 * PANEL_B, stream>>>(
        y_bf, wprojT, b_proj, (void*)out, nullptr, (int)M, C, C);
  }
}
